// CrossViewSwapAttention_1975684956350
// MI455X (gfx1250) — matrix operand order repaired and checked by reading, not compiled, not run on hardware
//
#include <hip/hip_runtime.h>
#include <hip/hip_bf16.h>
#include <math.h>

// ---------------------------------------------------------------------------
// CrossViewSwapAttention for MI455X (gfx1250, wave32, WMMA).
// All large GEMMs use v_wmma_f32_16x16x32_bf16. Param leaves are assumed
// flattened in recursive dict-insertion order of setup_inputs().
// ---------------------------------------------------------------------------

typedef __bf16 bf16;
typedef __attribute__((ext_vector_type(16))) __bf16 v16bf;
typedef __attribute__((ext_vector_type(8)))  float  v8f;
typedef int v4i __attribute__((vector_size(16)));

#define NVIEW   4
#define CDIM    128      // DIM
#define NHEAD   4
#define DHEAD   64
#define CINNER  256
#define FPIX    1024     // 32*32
#define BEV     128
#define NL      256      // windows
#define NT      256      // tokens per window
#define NCLUST  10

#if defined(__HIP_DEVICE_COMPILE__) && \
    __has_builtin(__builtin_amdgcn_global_load_async_to_lds_b128)
#define HAVE_ASYNC_LDS 1
typedef __attribute__((address_space(1))) v4i gv4i;
typedef __attribute__((address_space(3))) v4i lv4i;
#else
#define HAVE_ASYNC_LDS 0
#endif

static __device__ __forceinline__ v8f wmma_bf16(v16bf a, v16bf b, v8f c) {
#if defined(__HIP_DEVICE_COMPILE__)
  return __builtin_amdgcn_wmma_f32_16x16x32_bf16(false, a, false, b, (short)0, c,
                                                 false, false);
#else
  return c;
#endif
}

static __device__ __forceinline__ v8f vzero() {
  v8f z;
#pragma unroll
  for (int i = 0; i < 8; ++i) z[i] = 0.0f;
  return z;
}

union FragU { v16bf v; unsigned u[8]; };

// A fragment (16x32 bf16, M=rows striped over lanes 0..15, K split by lane-half):
// elem pair j<4: K = 2j + 8*half ; j>=4: K = 16 + 2(j-4) + 8*half  (ISA 7.12.2)
static __device__ __forceinline__ v16bf ldA(const bf16* base, int ld, int row0,
                                            int k0, int lane) {
  int m = lane & 15, h = lane >> 4;
  const bf16* r = base + (size_t)(row0 + m) * ld + k0 + h * 8;
  FragU f;
#pragma unroll
  for (int j = 0; j < 4; ++j) f.u[j] = *(const unsigned*)(r + 2 * j);
#pragma unroll
  for (int j = 0; j < 4; ++j) f.u[4 + j] = *(const unsigned*)(r + 16 + 2 * j);
  return f.v;
}

// B fragment (32x16 bf16). Source is stored N-major with K contiguous
// (i.e. Bt[n][k]): lane = column n (lane&15), K same order as ldA.
static __device__ __forceinline__ v16bf ldB(const bf16* baseT, int ldk, int n0,
                                            int k0, int lane) {
  int n = lane & 15, h = lane >> 4;
  const bf16* r = baseT + (size_t)(n0 + n) * ldk + k0 + h * 8;
  FragU f;
#pragma unroll
  for (int j = 0; j < 4; ++j) f.u[j] = *(const unsigned*)(r + 2 * j);
#pragma unroll
  for (int j = 0; j < 4; ++j) f.u[4 + j] = *(const unsigned*)(r + 16 + 2 * j);
  return f.v;
}

static __device__ __forceinline__ float half_max(float v) {
#pragma unroll
  for (int m = 1; m <= 8; m <<= 1) v = fmaxf(v, __shfl_xor(v, m, 32));
  return v;
}
static __device__ __forceinline__ float half_sum(float v) {
#pragma unroll
  for (int m = 1; m <= 8; m <<= 1) v += __shfl_xor(v, m, 32);
  return v;
}

// ---------------------------------------------------------------------------
__global__ void k_cembed(const float* __restrict__ E_inv,
                         const float* __restrict__ cam_w,
                         float* __restrict__ c_embed) {
  int i = threadIdx.x;
  if (i < NVIEW * CDIM) {
    int n = i >> 7, d = i & 127;
    float s = 0.f;
#pragma unroll
    for (int c = 0; c < 4; ++c)
      s += E_inv[(n * 4 + c) * 4 + 3] * cam_w[d * 4 + c];
    c_embed[i] = s;
  }
}

__global__ void k_imgembed(const float* __restrict__ I_inv,
                           const float* __restrict__ E_inv,
                           const float* __restrict__ img_w,
                           const float* __restrict__ image_plane,
                           const float* __restrict__ c_embed,
                           float* __restrict__ img_embed) {
  int bx = blockIdx.x;                 // n*1024 + p
  int n = bx >> 10, p = bx & 1023;
  int d = threadIdx.x;                 // 128 threads
  float px = image_plane[p], py = image_plane[1024 + p],
        pz = image_plane[2048 + p];
  float cam[4];
#pragma unroll
  for (int i = 0; i < 3; ++i)
    cam[i] = I_inv[(n * 3 + i) * 3 + 0] * px + I_inv[(n * 3 + i) * 3 + 1] * py +
             I_inv[(n * 3 + i) * 3 + 2] * pz;
  cam[3] = 1.0f;
  float dv[4];
#pragma unroll
  for (int i = 0; i < 4; ++i) {
    float s = 0.f;
#pragma unroll
    for (int j = 0; j < 4; ++j) s += E_inv[(n * 4 + i) * 4 + j] * cam[j];
    dv[i] = s;
  }
  float val = img_w[d * 4 + 0] * dv[0] + img_w[d * 4 + 1] * dv[1] +
              img_w[d * 4 + 2] * dv[2] + img_w[d * 4 + 3] * dv[3] -
              c_embed[n * CDIM + d];
  __shared__ float red[128];
  red[d] = val * val;
  __syncthreads();
#pragma unroll
  for (int s = 64; s > 0; s >>= 1) {
    if (d < s) red[d] += red[d + s];
    __syncthreads();
  }
  float denom = sqrtf(red[0]) + 1e-7f;
  img_embed[(size_t)(n * CDIM + d) * FPIX + p] = val / denom;
}

// BN+ReLU+conv1x1 (both weight sets) via WMMA.  key_flat = img_embed + X@Wp^T
// Two straight-line phases (no runtime branch in GEMM epilogues).
__global__ void k_featconv(const float* __restrict__ feature,
                           const float* __restrict__ pg, const float* __restrict__ pb,
                           const float* __restrict__ pm, const float* __restrict__ pv,
                           const float* __restrict__ pw,
                           const float* __restrict__ lg, const float* __restrict__ lb,
                           const float* __restrict__ lm, const float* __restrict__ lv,
                           const float* __restrict__ lw,
                           const float* __restrict__ img_embed,
                           float* __restrict__ key_flat,
                           float* __restrict__ val_flat) {
  __shared__ bf16 Xs[128 * 128];   // [pixel][c]
  __shared__ bf16 Wp[128 * 128];   // [d][c]
  __shared__ bf16 Wl[128 * 128];
  int bx = blockIdx.x;             // n*8 + ptile
  int n = bx >> 3, pbase = (bx & 7) * 128;
  int tid = threadIdx.x;
  int lane = tid & 31, wv = tid >> 5;

  // ---- phase 1: proj-branch BN + GEMM -> key_flat ----
  for (int e = tid; e < 128 * 128; e += 256) {
    int c = e >> 7, p = e & 127;
    float f = feature[(size_t)(n * CDIM + c) * FPIX + pbase + p];
    float t = (f - pm[c]) * rsqrtf(pv[c] + 1e-5f);
    t = fmaxf(t * pg[c] + pb[c], 0.0f);
    Xs[p * 128 + c] = (bf16)t;
    Wp[e] = (bf16)pw[e];
    Wl[e] = (bf16)lw[e];
  }
  __syncthreads();
  for (int i = 0; i < 8; ++i) {
    int cid = wv * 8 + i;
    int mt = cid >> 3, nt = cid & 7;
    v8f acc = vzero();
#pragma unroll
    for (int kc = 0; kc < 4; ++kc)
      acc = wmma_bf16(ldA(Xs, 128, mt * 16, kc * 32, lane),
                      ldB(Wp, 128, nt * 16, kc * 32, lane), acc);
#pragma unroll
    for (int r = 0; r < 8; ++r) {
      int row = mt * 16 + (lane >> 4) * 8 + r;
      int col = nt * 16 + (lane & 15);
      size_t idx = (size_t)(n * CDIM + col) * FPIX + pbase + row;
      key_flat[idx] = acc[r] + img_embed[idx];
    }
  }
  // ---- phase 2: lin-branch BN + GEMM -> val_flat ----
  __syncthreads();
  for (int e = tid; e < 128 * 128; e += 256) {
    int c = e >> 7, p = e & 127;
    float f = feature[(size_t)(n * CDIM + c) * FPIX + pbase + p];
    float t = (f - lm[c]) * rsqrtf(lv[c] + 1e-5f);
    t = fmaxf(t * lg[c] + lb[c], 0.0f);
    Xs[p * 128 + c] = (bf16)t;
  }
  __syncthreads();
  for (int i = 0; i < 8; ++i) {
    int cid = wv * 8 + i;
    int mt = cid >> 3, nt = cid & 7;
    v8f acc = vzero();
#pragma unroll
    for (int kc = 0; kc < 4; ++kc)
      acc = wmma_bf16(ldA(Xs, 128, mt * 16, kc * 32, lane),
                      ldB(Wl, 128, nt * 16, kc * 32, lane), acc);
#pragma unroll
    for (int r = 0; r < 8; ++r) {
      int row = mt * 16 + (lane >> 4) * 8 + r;
      int col = nt * 16 + (lane & 15);
      val_flat[(size_t)(n * CDIM + col) * FPIX + pbase + row] = acc[r];
    }
  }
}

__global__ void k_resize(const float* __restrict__ key_flat,
                         const float* __restrict__ val_flat,
                         bf16* __restrict__ keyup, bf16* __restrict__ valup) {
  int h = blockIdx.x, n = blockIdx.y, w = threadIdx.x;
  float fy = (h + 0.5f) * 0.25f - 0.5f;
  int iy = (int)floorf(fy);
  float wy = fy - (float)iy;
  int y0 = min(31, max(0, iy)), y1 = min(31, max(0, iy + 1));
  float fx = (w + 0.5f) * 0.25f - 0.5f;
  int ix = (int)floorf(fx);
  float wx = fx - (float)ix;
  int x0 = min(31, max(0, ix)), x1 = min(31, max(0, ix + 1));
  float w00 = (1.f - wy) * (1.f - wx), w01 = (1.f - wy) * wx;
  float w10 = wy * (1.f - wx), w11 = wy * wx;
  for (int d = 0; d < CDIM; ++d) {
    size_t base = (size_t)(n * CDIM + d) * FPIX;
    size_t o = ((size_t)(n * CDIM + d) * BEV + h) * BEV + w;
    float k = key_flat[base + y0 * 32 + x0] * w00 + key_flat[base + y0 * 32 + x1] * w01 +
              key_flat[base + y1 * 32 + x0] * w10 + key_flat[base + y1 * 32 + x1] * w11;
    float v = val_flat[base + y0 * 32 + x0] * w00 + val_flat[base + y0 * 32 + x1] * w01 +
              val_flat[base + y1 * 32 + x0] * w10 + val_flat[base + y1 * 32 + x1] * w11;
    keyup[o] = (bf16)k;
    valup[o] = (bf16)v;
  }
}

__global__ void k_query(const float* __restrict__ learned,
                        const float* __restrict__ grid0,
                        const float* __restrict__ bev_w,
                        const float* __restrict__ bev_b,
                        const float* __restrict__ c_embed,
                        bf16* __restrict__ query) {
  __shared__ float cmean[128 * 128];   // [d][w]
  int h = blockIdx.x, w = threadIdx.x;
  float g0 = grid0[h * BEV + w];
  float g1 = grid0[BEV * BEV + h * BEV + w];
  float n2[NVIEW] = {0.f, 0.f, 0.f, 0.f};
  for (int d = 0; d < CDIM; ++d) {
    float cm = 0.f;
#pragma unroll
    for (int k = 0; k < NCLUST; ++k)
      cm += learned[((size_t)(k * CDIM + d) * BEV + h) * BEV + w];
    cm *= 0.1f;
    cmean[d * 128 + w] = cm;
    float we = g0 * bev_w[d * 2] + g1 * bev_w[d * 2 + 1] + bev_b[d];
#pragma unroll
    for (int n = 0; n < NVIEW; ++n) {
      float diff = we - c_embed[n * CDIM + d];
      n2[n] += diff * diff;
    }
  }
  float rinv[NVIEW];
#pragma unroll
  for (int n = 0; n < NVIEW; ++n) rinv[n] = 1.f / (sqrtf(n2[n]) + 1e-7f);
  for (int d = 0; d < CDIM; ++d) {
    float we = g0 * bev_w[d * 2] + g1 * bev_w[d * 2 + 1] + bev_b[d];
    float cm = cmean[d * 128 + w];
#pragma unroll
    for (int n = 0; n < NVIEW; ++n) {
      float q = (we - c_embed[n * CDIM + d]) * rinv[n] + cm;
      query[((size_t)(n * CDIM + d) * BEV + h) * BEV + w] = (bf16)q;
    }
  }
}

// token -> (view n, pixel h,w).  MAP==0: window 8x8, MAP==1: dilated grid.
template <int MAP>
static __device__ __forceinline__ void tok2pix(int l, int t, int& n, int& hh,
                                               int& ww) {
  int x = l >> 4, y = l & 15;
  n = t >> 6;
  int inner = t & 63, a = inner >> 3, c = inner & 7;
  if (MAP == 0) { hh = x * 8 + a;  ww = y * 8 + c; }
  else          { hh = a * 16 + x; ww = c * 16 + y; }
}

// LayerNorm + projection 128->256 (WMMA).  SRCKIND 0: bf16 [n][d][h][w],
// SRCKIND 1: f32 token-major [h][w][d] broadcast over n.
template <int SRCKIND, int MAP>
__global__ void k_proj(bf16* __restrict__ dst, const void* __restrict__ src,
                       const float* __restrict__ ln_g,
                       const float* __restrict__ ln_b,
                       const float* __restrict__ W,
                       const float* __restrict__ bias) {
  __shared__ bf16 A[NT * 128];       // LN'd tokens
  __shared__ bf16 Wt[CINNER * 128];  // [e][din]
  int l = blockIdx.x, tid = threadIdx.x;
  for (int e = tid; e < CINNER * 128; e += 256) {
    int eo = e >> 7, k = e & 127;
    Wt[e] = (bf16)W[k * CINNER + eo];
  }
  {
    int t = tid, n, hh, ww;
    tok2pix<MAP>(l, t, n, hh, ww);
    float s = 0.f, s2 = 0.f;
    for (int d = 0; d < CDIM; ++d) {
      float v;
      if (SRCKIND == 0)
        v = (float)((const bf16*)src)[((size_t)(n * CDIM + d) * BEV + hh) * BEV + ww];
      else
        v = ((const float*)src)[((size_t)hh * BEV + ww) * CDIM + d];
      s += v; s2 += v * v;
      A[t * 128 + d] = (bf16)v;
    }
    float mu = s * (1.f / 128.f);
    float var = s2 * (1.f / 128.f) - mu * mu;
    float rs = rsqrtf(var + 1e-5f);
    for (int d = 0; d < CDIM; ++d) {
      float v = ((float)A[t * 128 + d] - mu) * rs * ln_g[d] + ln_b[d];
      A[t * 128 + d] = (bf16)v;
    }
  }
  __syncthreads();
  int lane = tid & 31, wv = tid >> 5;
  for (int i = 0; i < 32; ++i) {
    int cid = wv * 32 + i;
    int mt = cid >> 4, nt = cid & 15;
    v8f acc = vzero();
#pragma unroll
    for (int kc = 0; kc < 4; ++kc)
      acc = wmma_bf16(ldA(A, 128, mt * 16, kc * 32, lane),
                      ldB(Wt, 128, nt * 16, kc * 32, lane), acc);
#pragma unroll
    for (int r = 0; r < 8; ++r) {
      int row = mt * 16 + (lane >> 4) * 8 + r;
      int col = nt * 16 + (lane & 15);
      dst[((size_t)l * NT + row) * CINNER + col] = (bf16)(acc[r] + bias[col]);
    }
  }
}

// Flash attention core: block = (window, head), 8 waves x 32 query rows.
__global__ void k_flash(bf16* __restrict__ O, const bf16* __restrict__ Q,
                        const bf16* __restrict__ K, const bf16* __restrict__ V) {
  __shared__ bf16 Qs[NT * DHEAD];
  __shared__ bf16 Ks[NT * DHEAD];
  __shared__ bf16 Vt[DHEAD * NT];
  __shared__ bf16 Pscr[8 * 32 * 32];
  int l = blockIdx.x, m = blockIdx.y, tid = threadIdx.x;
  size_t base = (size_t)l * NT * CINNER + m * DHEAD;
#if HAVE_ASYNC_LDS
  // Async DMA (tracked by ASYNCcnt) for the contiguous 128B row segments.
  for (int e = tid; e < 2048; e += 256) {
    int t = e >> 3, seg = e & 7;
    __builtin_amdgcn_global_load_async_to_lds_b128(
        (gv4i*)(void*)(Q + base + (size_t)t * CINNER + seg * 8),
        (lv4i*)(void*)&Qs[t * DHEAD + seg * 8], 0, 0);
    __builtin_amdgcn_global_load_async_to_lds_b128(
        (gv4i*)(void*)(K + base + (size_t)t * CINNER + seg * 8),
        (lv4i*)(void*)&Ks[t * DHEAD + seg * 8], 0, 0);
  }
#else
  for (int e = tid; e < NT * DHEAD; e += 256) {
    int t = e >> 6, dh = e & 63;
    Qs[e] = Q[base + (size_t)t * CINNER + dh];
    Ks[e] = K[base + (size_t)t * CINNER + dh];
  }
#endif
  for (int e = tid; e < NT * DHEAD; e += 256) {
    int dh = e >> 8, t = e & 255;
    Vt[dh * NT + t] = V[base + (size_t)t * CINNER + dh];
  }
#if HAVE_ASYNC_LDS
#if __has_builtin(__builtin_amdgcn_s_wait_asynccnt)
  __builtin_amdgcn_s_wait_asynccnt(0);
#else
  asm volatile("s_wait_asynccnt 0" ::: "memory");
#endif
#endif
  __syncthreads();
  int lane = tid & 31, wv = tid >> 5, hl = lane >> 4;
  int qbase = wv * 32;
  bf16* Pw = &Pscr[wv * 32 * 32];
  float mrow[2][8], lrow[2][8];
  v8f accO[2][4];
#pragma unroll
  for (int s = 0; s < 2; ++s) {
#pragma unroll
    for (int r = 0; r < 8; ++r) { mrow[s][r] = -1e30f; lrow[s][r] = 0.f; }
#pragma unroll
    for (int nt = 0; nt < 4; ++nt) accO[s][nt] = vzero();
  }
  const float scale = 0.125f;   // DIM_HEAD^-0.5
  for (int kc = 0; kc < 8; ++kc) {
    v8f S[2][2];
#pragma unroll
    for (int s = 0; s < 2; ++s)
#pragma unroll
      for (int ntk = 0; ntk < 2; ++ntk) {
        v8f c = vzero();
#pragma unroll
        for (int kk = 0; kk < 2; ++kk)
          c = wmma_bf16(ldA(Qs, DHEAD, qbase + s * 16, kk * 32, lane),
                        ldB(Ks, DHEAD, kc * 32 + ntk * 16, kk * 32, lane), c);
        S[s][ntk] = c;
      }
#pragma unroll
    for (int s = 0; s < 2; ++s)
#pragma unroll
      for (int r = 0; r < 8; ++r) {
        float t0 = fmaxf(S[s][0][r], S[s][1][r]) * scale;
        float rmax = half_max(t0);
        float mo = mrow[s][r];
        float mn = fmaxf(mo, rmax);
        float alpha = __expf(mo - mn);
        mrow[s][r] = mn;
        float p0 = __expf(S[s][0][r] * scale - mn);
        float p1 = __expf(S[s][1][r] * scale - mn);
        lrow[s][r] = lrow[s][r] * alpha + half_sum(p0 + p1);
#pragma unroll
        for (int nt = 0; nt < 4; ++nt) accO[s][nt][r] *= alpha;
        int rl = s * 16 + hl * 8 + r;
        int cl = lane & 15;
        Pw[rl * 32 + cl] = (bf16)p0;
        Pw[rl * 32 + 16 + cl] = (bf16)p1;
      }
#if defined(__HIP_DEVICE_COMPILE__)
    asm volatile("s_wait_dscnt 0" ::: "memory");
#endif
#pragma unroll
    for (int s = 0; s < 2; ++s) {
      v16bf a = ldA(Pw, 32, s * 16, 0, lane);
#pragma unroll
      for (int nt = 0; nt < 4; ++nt)
        accO[s][nt] = wmma_bf16(a, ldB(Vt, NT, nt * 16, kc * 32, lane), accO[s][nt]);
    }
  }
#pragma unroll
  for (int s = 0; s < 2; ++s)
#pragma unroll
    for (int r = 0; r < 8; ++r) {
      float inv = 1.0f / lrow[s][r];
      int row = s * 16 + hl * 8 + r;
      int t = qbase + row;
#pragma unroll
      for (int nt = 0; nt < 4; ++nt) {
        int col = m * DHEAD + nt * 16 + (lane & 15);
        O[((size_t)l * NT + t) * CINNER + col] = (bf16)(accO[s][nt][r] * inv);
      }
    }
}

// Out-projection 256->128; mean over views folded into the WMMA C-chain.
template <int SKIPKIND, int MAP>
__global__ void k_outproj(float* __restrict__ q2d, const bf16* __restrict__ O,
                          const float* __restrict__ wo,
                          const float* __restrict__ bo,
                          const float* __restrict__ skip) {
  __shared__ bf16 WoT[CDIM * CINNER];  // [d][e]
  int l = blockIdx.x, tid = threadIdx.x;
  for (int e = tid; e < CDIM * CINNER; e += 256) {
    int d = e >> 8, eo = e & 255;
    WoT[e] = (bf16)wo[eo * CDIM + d];
  }
  __syncthreads();
  const bf16* Ob = O + (size_t)l * NT * CINNER;
  int lane = tid & 31, wv = tid >> 5;
  int q = wv & 3, nh = wv >> 2;
  int x = l >> 4, y = l & 15;
  for (int j = 0; j < 4; ++j) {
    int nt = nh * 4 + j;
    v8f acc = vzero();
    for (int n = 0; n < NVIEW; ++n)
#pragma unroll
      for (int kc = 0; kc < 8; ++kc)
        acc = wmma_bf16(ldA(Ob, CINNER, n * 64 + q * 16, kc * 32, lane),
                        ldB(WoT, CINNER, nt * 16, kc * 32, lane), acc);
#pragma unroll
    for (int r = 0; r < 8; ++r) {
      int row = (lane >> 4) * 8 + r;
      int inner = q * 16 + row;
      int a = inner >> 3, bb = inner & 7;
      int hh, ww;
      if (MAP == 0) { hh = x * 8 + a;  ww = y * 8 + bb; }
      else          { hh = a * 16 + x; ww = bb * 16 + y; }
      int col = nt * 16 + (lane & 15);
      float sk = (SKIPKIND == 0)
                     ? skip[(size_t)col * (BEV * BEV) + hh * BEV + ww]
                     : skip[((size_t)hh * BEV + ww) * CDIM + col];
      q2d[((size_t)hh * BEV + ww) * CDIM + col] = acc[r] * 0.25f + bo[col] + sk;
    }
  }
}

// MLP stage 1: LN + 128->256 GEMM + exact GELU
__global__ void k_mlp1(bf16* __restrict__ H, const float* __restrict__ q2d,
                       const float* __restrict__ ln_g, const float* __restrict__ ln_b,
                       const float* __restrict__ w1, const float* __restrict__ b1) {
  __shared__ bf16 A[128 * 128];
  __shared__ bf16 W1t[CINNER * 128];  // [e][k]
  int tbase = blockIdx.x * 128, tid = threadIdx.x;
  for (int e = tid; e < CINNER * 128; e += 256) {
    int eo = e >> 7, k = e & 127;
    W1t[e] = (bf16)w1[k * CINNER + eo];
  }
  if (tid < 128) {
    int t = tid;
    const float* row = q2d + (size_t)(tbase + t) * CDIM;
    float s = 0.f, s2 = 0.f;
    for (int d = 0; d < CDIM; ++d) { float v = row[d]; s += v; s2 += v * v; }
    float mu = s * (1.f / 128.f);
    float var = s2 * (1.f / 128.f) - mu * mu;
    float rs = rsqrtf(var + 1e-5f);
    for (int d = 0; d < CDIM; ++d)
      A[t * 128 + d] = (bf16)((row[d] - mu) * rs * ln_g[d] + ln_b[d]);
  }
  __syncthreads();
  int lane = tid & 31, wv = tid >> 5;
  for (int i = 0; i < 16; ++i) {
    int cid = wv * 16 + i;
    int mt = cid >> 4, nt = cid & 15;
    v8f acc = vzero();
#pragma unroll
    for (int kc = 0; kc < 4; ++kc)
      acc = wmma_bf16(ldA(A, 128, mt * 16, kc * 32, lane),
                      ldB(W1t, 128, nt * 16, kc * 32, lane), acc);
#pragma unroll
    for (int r = 0; r < 8; ++r) {
      int row = mt * 16 + (lane >> 4) * 8 + r;
      int col = nt * 16 + (lane & 15);
      float vv = acc[r] + b1[col];
      float ge = 0.5f * vv * (1.f + erff(vv * 0.70710678f));
      H[(size_t)(tbase + row) * CINNER + col] = (bf16)ge;
    }
  }
}

// MLP stage 2: 256->128 GEMM + residual (in place on q2d)
__global__ void k_mlp2(float* __restrict__ q2d, const bf16* __restrict__ H,
                       const float* __restrict__ w2, const float* __restrict__ b2) {
  __shared__ bf16 W2t[CDIM * CINNER];  // [d][e]
  int tbase = blockIdx.x * 128, tid = threadIdx.x;
  for (int e = tid; e < CDIM * CINNER; e += 256) {
    int d = e >> 8, eo = e & 255;
    W2t[e] = (bf16)w2[eo * CDIM + d];
  }
  __syncthreads();
  const bf16* Hb = H + (size_t)tbase * CINNER;
  int lane = tid & 31, wv = tid >> 5;
  for (int i = 0; i < 8; ++i) {
    int cid = wv * 8 + i;
    int mt = cid >> 3, nt = cid & 7;
    v8f acc = vzero();
#pragma unroll
    for (int kc = 0; kc < 8; ++kc)
      acc = wmma_bf16(ldA(Hb, CINNER, mt * 16, kc * 32, lane),
                      ldB(W2t, CINNER, nt * 16, kc * 32, lane), acc);
#pragma unroll
    for (int r = 0; r < 8; ++r) {
      int row = mt * 16 + (lane >> 4) * 8 + r;
      int col = nt * 16 + (lane & 15);
      size_t idx = (size_t)(tbase + row) * CDIM + col;
      q2d[idx] = q2d[idx] + acc[r] + b2[col];
    }
  }
}

__global__ void k_final(float* __restrict__ out, const float* __restrict__ q2d,
                        const float* __restrict__ g, const float* __restrict__ b) {
  int h = blockIdx.x, w = threadIdx.x;
  const float* row = q2d + ((size_t)h * BEV + w) * CDIM;
  float s = 0.f, s2 = 0.f;
  for (int d = 0; d < CDIM; ++d) { float v = row[d]; s += v; s2 += v * v; }
  float mu = s * (1.f / 128.f);
  float var = s2 * (1.f / 128.f) - mu * mu;
  float rs = rsqrtf(var + 1e-5f);
  for (int d = 0; d < CDIM; ++d)
    out[(size_t)d * (BEV * BEV) + h * BEV + w] = (row[d] - mu) * rs * g[d] + b[d];
}

// ---------------------------------------------------------------------------
extern "C" void kernel_launch(void* const* d_in, const int* in_sizes, int n_in,
                              void* d_out, int out_size, void* d_ws, size_t ws_size,
                              hipStream_t stream) {
  (void)in_sizes; (void)n_in; (void)out_size; (void)ws_size;
  const float* x        = (const float*)d_in[0];
  const float* feature  = (const float*)d_in[1];
  const float* I_inv    = (const float*)d_in[2];
  const float* E_inv    = (const float*)d_in[3];
  // d_in[4] cluster_ids, d_in[5] index : unused by the reference math
  const float* cam_w    = (const float*)d_in[6];
  const float* img_w    = (const float*)d_in[7];
  const float* bev_w    = (const float*)d_in[8];
  const float* bev_b    = (const float*)d_in[9];
  const float* fp_g = (const float*)d_in[10], *fp_b = (const float*)d_in[11];
  const float* fp_m = (const float*)d_in[12], *fp_v = (const float*)d_in[13];
  const float* fp_w = (const float*)d_in[14];
  const float* fl_g = (const float*)d_in[15], *fl_b = (const float*)d_in[16];
  const float* fl_m = (const float*)d_in[17], *fl_v = (const float*)d_in[18];
  const float* fl_w = (const float*)d_in[19];
  const float* a1[14]; for (int i = 0; i < 14; ++i) a1[i] = (const float*)d_in[20 + i];
  const float* a2[14]; for (int i = 0; i < 14; ++i) a2[i] = (const float*)d_in[34 + i];
  const float* pre1_g = (const float*)d_in[48], *pre1_b = (const float*)d_in[49];
  const float* pre2_g = (const float*)d_in[50], *pre2_b = (const float*)d_in[51];
  const float* post_g = (const float*)d_in[52], *post_b = (const float*)d_in[53];
  const float* m1w1 = (const float*)d_in[54], *m1b1 = (const float*)d_in[55];
  const float* m1w2 = (const float*)d_in[56], *m1b2 = (const float*)d_in[57];
  const float* m2w1 = (const float*)d_in[58], *m2b1 = (const float*)d_in[59];
  const float* m2w2 = (const float*)d_in[60], *m2b2 = (const float*)d_in[61];
  const float* learned     = (const float*)d_in[62];
  const float* grid0       = (const float*)d_in[63];
  const float* image_plane = (const float*)d_in[64];

  char* w = (char*)d_ws;
  size_t off = 0;
  auto alloc = [&](size_t bytes) {
    void* p = w + off;
    off = (off + bytes + 255) & ~(size_t)255;
    return p;
  };
  float* c_embed  = (float*)alloc(512 * 4);
  float* img_emb  = (float*)alloc((size_t)4 * 128 * 1024 * 4);
  float* key_flat = (float*)alloc((size_t)4 * 128 * 1024 * 4);
  float* val_flat = (float*)alloc((size_t)4 * 128 * 1024 * 4);
  bf16*  keyup    = (bf16*)alloc((size_t)4 * 128 * 128 * 128 * 2);
  bf16*  valup    = (bf16*)alloc((size_t)4 * 128 * 128 * 128 * 2);
  bf16*  query    = (bf16*)alloc((size_t)4 * 128 * 128 * 128 * 2);
  bf16*  Qh       = (bf16*)alloc((size_t)NL * NT * CINNER * 2);
  bf16*  Kh       = (bf16*)alloc((size_t)NL * NT * CINNER * 2);
  bf16*  Vh       = (bf16*)alloc((size_t)NL * NT * CINNER * 2);
  bf16*  Obuf     = (bf16*)alloc((size_t)NL * NT * CINNER * 2);
  float* q2dA     = (float*)alloc((size_t)128 * 128 * 128 * 4);
  float* q2dB     = (float*)alloc((size_t)128 * 128 * 128 * 4);
  bf16*  Hbuf     = (bf16*)alloc((size_t)16384 * 256 * 2);

  k_cembed<<<1, 512, 0, stream>>>(E_inv, cam_w, c_embed);
  k_imgembed<<<4096, 128, 0, stream>>>(I_inv, E_inv, img_w, image_plane,
                                       c_embed, img_emb);
  k_featconv<<<32, 256, 0, stream>>>(feature, fp_g, fp_b, fp_m, fp_v, fp_w,
                                     fl_g, fl_b, fl_m, fl_v, fl_w, img_emb,
                                     key_flat, val_flat);
  k_resize<<<dim3(128, 4), 128, 0, stream>>>(key_flat, val_flat, keyup, valup);
  k_query<<<128, 128, 0, stream>>>(learned, grid0, bev_w, bev_b, c_embed, query);

  // ---- attention block 1 (window partition) ----
  k_proj<0, 0><<<NL, 256, 0, stream>>>(Qh, query, a1[0], a1[1], a1[6], a1[7]);
  k_proj<0, 0><<<NL, 256, 0, stream>>>(Kh, keyup, a1[2], a1[3], a1[8], a1[9]);
  k_proj<0, 0><<<NL, 256, 0, stream>>>(Vh, valup, a1[4], a1[5], a1[10], a1[11]);
  k_flash<<<dim3(NL, NHEAD), 256, 0, stream>>>(Obuf, Qh, Kh, Vh);
  k_outproj<0, 0><<<NL, 256, 0, stream>>>(q2dA, Obuf, a1[12], a1[13], x);
  k_mlp1<<<128, 256, 0, stream>>>(Hbuf, q2dA, pre1_g, pre1_b, m1w1, m1b1);
  k_mlp2<<<128, 256, 0, stream>>>(q2dA, Hbuf, m1w2, m1b2);

  // ---- attention block 2 (dilated grid partition) ----
  k_proj<1, 1><<<NL, 256, 0, stream>>>(Qh, q2dA, a2[0], a2[1], a2[6], a2[7]);
  k_proj<0, 1><<<NL, 256, 0, stream>>>(Kh, keyup, a2[2], a2[3], a2[8], a2[9]);
  k_proj<0, 1><<<NL, 256, 0, stream>>>(Vh, valup, a2[4], a2[5], a2[10], a2[11]);
  k_flash<<<dim3(NL, NHEAD), 256, 0, stream>>>(Obuf, Qh, Kh, Vh);
  k_outproj<1, 1><<<NL, 256, 0, stream>>>(q2dB, Obuf, a2[12], a2[13], q2dA);
  k_mlp1<<<128, 256, 0, stream>>>(Hbuf, q2dB, pre2_g, pre2_b, m2w1, m2b1);
  k_mlp2<<<128, 256, 0, stream>>>(q2dB, Hbuf, m2w2, m2b2);

  k_final<<<128, 128, 0, stream>>>((float*)d_out, q2dB, post_g, post_b);
}
